// similar_RNN_multi_52853867544801
// MI455X (gfx1250) — matrix operand order repaired and checked by reading, not compiled, not run on hardware
//
#include <hip/hip_runtime.h>
#include <hip/hip_bf16.h>
#include <math.h>

// Problem constants (match reference)
constexpr int B = 32;
constexpr int T = 1024;
constexpr int D = 128;     // x feature dim; seq dim = 2*D = 256
constexpr int D2 = 256;    // 2*D
constexpr int HEADS = 4;
constexpr int HP = 16;     // heads padded to 16 WMMA N-columns
constexpr int UNITS = 128;
constexpr int F = HEADS * UNITS; // 512
constexpr float DEC = 0.85f;
constexpr float OMD = 0.15f;

// Scan chunking: exact linear-recurrence decomposition (no approximation).
constexpr int CCH = 16;          // chunks per batch
constexpr int CLEN = T / CCH;    // 64 timesteps per chunk
// c-pass chunking
constexpr int SC = 8;            // s-chunks per batch
constexpr int SLEN = T / SC;     // 128 rows per chunk

typedef __attribute__((ext_vector_type(16))) _Float16 v16h;
typedef __attribute__((ext_vector_type(8)))  float    v8f;

// ---------------------------------------------------------------------------
// 1a) Scan pass A: per-chunk local scan from zero state; record chunk-final s1.
//     grid(B*CCH), block(D). Serial depth = CLEN = 64.
// ---------------------------------------------------------------------------
__global__ void scan_partial_kernel(const float* __restrict__ x,
                                    float* __restrict__ locend) {
    const int b = blockIdx.x / CCH;
    const int c = blockIdx.x % CCH;
    const int d = threadIdx.x;
    const float* xp = x + ((size_t)b * T + c * CLEN) * D + d;
    float s1 = 0.0f;
    for (int t = 0; t < CLEN; t += 4) {
        float x0 = xp[(size_t)(t + 0) * D];
        float x1 = xp[(size_t)(t + 1) * D];
        float x2 = xp[(size_t)(t + 2) * D];
        float x3 = xp[(size_t)(t + 3) * D];
        s1 = DEC * s1 + OMD * x0;
        s1 = DEC * s1 + OMD * x1;
        s1 = DEC * s1 + OMD * x2;
        s1 = DEC * s1 + OMD * x3;
    }
    locend[((size_t)b * CCH + c) * D + d] = s1;
}

// ---------------------------------------------------------------------------
// 1b) Scan pass B: combine carries. s1_start[c] = DEC^CLEN * s1_start[c-1] +
//     locend[c-1]. grid(B), block(D); 16-long serial loop (trivial).
// ---------------------------------------------------------------------------
__global__ void scan_combine_kernel(const float* __restrict__ locend,
                                    float* __restrict__ init) {
    const int b = blockIdx.x;
    const int d = threadIdx.x;
    float dl = 1.0f;
    #pragma unroll
    for (int i = 0; i < CLEN; ++i) dl *= DEC;   // DEC^64 (exact fp32 product)
    float s = 0.0f;
    for (int c = 0; c < CCH; ++c) {
        init[((size_t)b * CCH + c) * D + d] = s;
        s = dl * s + locend[((size_t)b * CCH + c) * D + d];
    }
}

// ---------------------------------------------------------------------------
// 1c) Scan pass C: re-run each chunk from its exact carried start, write seq.
//     grid(B*CCH), block(D). Serial depth = 64.
// ---------------------------------------------------------------------------
__global__ void scan_write_kernel(const float* __restrict__ x,
                                  const float* __restrict__ init,
                                  float* __restrict__ seq) {
    const int b = blockIdx.x / CCH;
    const int c = blockIdx.x % CCH;
    const int d = threadIdx.x;
    const float* xp = x + ((size_t)b * T + c * CLEN) * D + d;
    float* sp = seq + ((size_t)b * T + c * CLEN) * D2 + d;
    float s1 = init[((size_t)b * CCH + c) * D + d];
    for (int t = 0; t < CLEN; t += 4) {
        float x0 = xp[(size_t)(t + 0) * D];
        float x1 = xp[(size_t)(t + 1) * D];
        float x2 = xp[(size_t)(t + 2) * D];
        float x3 = xp[(size_t)(t + 3) * D];
        if (t + 16 < CLEN) __builtin_prefetch(xp + (size_t)(t + 16) * D, 0, 0);
        float ns;
        ns = DEC * s1 + OMD * x0;
        sp[(size_t)(t + 0) * D2]     = ns;
        sp[(size_t)(t + 0) * D2 + D] = OMD * s1 + DEC * x0;
        s1 = ns;
        ns = DEC * s1 + OMD * x1;
        sp[(size_t)(t + 1) * D2]     = ns;
        sp[(size_t)(t + 1) * D2 + D] = OMD * s1 + DEC * x1;
        s1 = ns;
        ns = DEC * s1 + OMD * x2;
        sp[(size_t)(t + 2) * D2]     = ns;
        sp[(size_t)(t + 2) * D2 + D] = OMD * s1 + DEC * x2;
        s1 = ns;
        ns = DEC * s1 + OMD * x3;
        sp[(size_t)(t + 3) * D2]     = ns;
        sp[(size_t)(t + 3) * D2 + D] = OMD * s1 + DEC * x3;
        s1 = ns;
    }
}

// ---------------------------------------------------------------------------
// 2) q_last[b,h,u] = sum_d seq[b,T-1,d] * qk[d,h,u]
//    wk[b,h,d]     = sum_u q_last[b,h,u] * kk[d,h,u]
//    Stored head-padded: wkp[B][16][256] (h >= HEADS rows zeroed for WMMA pad).
//    grid(B*16), block(128)
// ---------------------------------------------------------------------------
__global__ void qwk_kernel(const float* __restrict__ seq,
                           const float* __restrict__ qk,
                           const float* __restrict__ kk,
                           float* __restrict__ wkp) {
    const int b = blockIdx.x >> 4;
    const int h = blockIdx.x & 15;
    const int tid = threadIdx.x; // 0..127
    float* wrow = wkp + ((size_t)b * HP + h) * D2;

    if (h >= HEADS) {
        wrow[tid] = 0.0f;
        wrow[tid + 128] = 0.0f;
        return;
    }

    __shared__ float sl[D2];   // seq[b, T-1, :]
    __shared__ float ql[UNITS];

    const float* slast = seq + ((size_t)b * T + (T - 1)) * D2;
    sl[tid]       = slast[tid];
    sl[tid + 128] = slast[tid + 128];
    __syncthreads();

    float q = 0.0f;
    #pragma unroll 4
    for (int d = 0; d < D2; ++d)
        q += sl[d] * qk[(size_t)d * F + h * UNITS + tid];
    ql[tid] = q;
    __syncthreads();

    #pragma unroll
    for (int i = 0; i < 2; ++i) {
        const int d2 = tid + i * 128;
        const float* krow = kk + (size_t)d2 * F + h * UNITS;
        float w = 0.0f;
        #pragma unroll 4
        for (int u = 0; u < UNITS; ++u)
            w += ql[u] * krow[u];
        wrow[d2] = w;
    }
}

// ---------------------------------------------------------------------------
// 3) scores[b,h,s] = (1/sqrt(128)) * sum_d seq[b,s,d] * wk[b,h,d]
//    via v_wmma_f32_16x16x32_f16. grid(B*8), block(32): one wave, 8 row-tiles.
// ---------------------------------------------------------------------------
__global__ void score_wmma_kernel(const float* __restrict__ seq,
                                  const float* __restrict__ wkp,
                                  float* __restrict__ sco) {
    const int b    = blockIdx.x >> 3;
    const int w    = blockIdx.x & 7;
    const int lane = threadIdx.x;      // 0..31
    const int M    = lane & 15;
    const int half = lane >> 4;        // 0 or 1
    const int N    = lane & 15;        // column index for B / D

    const float* wrow = wkp + ((size_t)b * HP + N) * D2;
    constexpr float SCALE = 0.08838834764831845f; // 1/sqrt(128)

    for (int ti = 0; ti < 8; ++ti) {
        const int tile = w * 8 + ti;          // 0..63
        const int srow = tile * 16 + M;
        const float* arow = seq + ((size_t)b * T + srow) * D2;

        v8f acc = {};
        #pragma unroll
        for (int kc = 0; kc < 8; ++kc) {
            const int k0 = kc * 32;
            // A fragment: lanes 0-15 -> K = 0..15, lanes 16-31 -> K = 16..31
            v16h af = {};
            const float* ap0 = arow + k0 + half * 16;
            const float* ap1 = arow + k0 + 8 + half * 16;
            #pragma unroll
            for (int e = 0; e < 8; ++e) {
                af[e]     = (_Float16)ap0[e];
                af[e + 8] = (_Float16)ap1[e];
            }
            // B fragment: lanes 0-15 -> K=0..15, lanes 16-31 -> K=16..31 of col N
            v16h bf = {};
            const float* bp = wrow + k0 + half * 16;
            #pragma unroll
            for (int e = 0; e < 16; ++e)
                bf[e] = (_Float16)bp[e];

            acc = __builtin_amdgcn_wmma_f32_16x16x32_f16(
                false, af, false, bf, (short)0, acc, false, false);
        }

        // D layout: VGPR r -> M = r + half*8, N = lane%16
        if (N < HEADS) {
            float* srow_out = sco + ((size_t)b * HEADS + N) * T + tile * 16;
            #pragma unroll
            for (int r = 0; r < 8; ++r)
                srow_out[r + half * 8] = acc[r] * SCALE;
        }
    }
}

// ---------------------------------------------------------------------------
// 4a) softmax over s per (b,h): probs[B][4][1024] to workspace.
//     grid(B), block(256)
// ---------------------------------------------------------------------------
__global__ void softmax_kernel(const float* __restrict__ sco,
                               float* __restrict__ probs) {
    const int b = blockIdx.x;
    const int tid = threadIdx.x; // 0..255
    __shared__ float red[256];

    for (int h = 0; h < HEADS; ++h) {
        const float* sc = sco + ((size_t)b * HEADS + h) * T;
        float* pr = probs + ((size_t)b * HEADS + h) * T;
        // max
        float m = -1e30f;
        for (int s = tid; s < T; s += 256) m = fmaxf(m, sc[s]);
        red[tid] = m; __syncthreads();
        for (int off = 128; off > 0; off >>= 1) {
            if (tid < off) red[tid] = fmaxf(red[tid], red[tid + off]);
            __syncthreads();
        }
        const float mx = red[0]; __syncthreads();
        // exp + sum
        float part = 0.0f;
        float ev[T / 256];
        #pragma unroll
        for (int i = 0; i < T / 256; ++i) {
            ev[i] = __expf(sc[tid + i * 256] - mx);
            part += ev[i];
        }
        red[tid] = part; __syncthreads();
        for (int off = 128; off > 0; off >>= 1) {
            if (tid < off) red[tid] += red[tid + off];
            __syncthreads();
        }
        const float inv = 1.0f / red[0]; __syncthreads();
        #pragma unroll
        for (int i = 0; i < T / 256; ++i) pr[tid + i * 256] = ev[i] * inv;
    }
}

// ---------------------------------------------------------------------------
// 4b) c partials: c_part[b][sc][h][d] = sum_{s in chunk} probs[h][s]*seq[b,s,d]
//     grid(B*SC), block(256) — 256 blocks streaming seq from L2.
// ---------------------------------------------------------------------------
__global__ void c_partial_kernel(const float* __restrict__ seq,
                                 const float* __restrict__ probs,
                                 float* __restrict__ c_part) {
    const int b = blockIdx.x / SC;
    const int sc = blockIdx.x % SC;
    const int tid = threadIdx.x; // 0..255

    __shared__ float pl[HEADS][SLEN]; // 2 KB
    // load probs chunk: 4*128 = 512 values, 2 per thread
    #pragma unroll
    for (int i = 0; i < 2; ++i) {
        const int idx = tid + i * 256;        // 0..511
        const int h = idx / SLEN;
        const int s = idx % SLEN;
        pl[h][s] = probs[((size_t)b * HEADS + h) * T + sc * SLEN + s];
    }
    __syncthreads();

    const int d = tid;
    float a0 = 0.f, a1 = 0.f, a2 = 0.f, a3 = 0.f;
    const float* sp = seq + ((size_t)b * T + sc * SLEN) * D2 + d;
    for (int s = 0; s < SLEN; ++s) {
        const float sv = sp[(size_t)s * D2];
        a0 += pl[0][s] * sv;
        a1 += pl[1][s] * sv;
        a2 += pl[2][s] * sv;
        a3 += pl[3][s] * sv;
    }
    float* cp = c_part + (((size_t)b * SC + sc) * HEADS) * D2 + d;
    cp[0 * D2] = a0;
    cp[1 * D2] = a1;
    cp[2 * D2] = a2;
    cp[3 * D2] = a3;
}

// ---------------------------------------------------------------------------
// 4c) deterministic reduction of c partials. grid(B), block(256)
// ---------------------------------------------------------------------------
__global__ void c_reduce_kernel(const float* __restrict__ c_part,
                                float* __restrict__ cvec) {
    const int b = blockIdx.x;
    const int d = threadIdx.x; // 0..255
    #pragma unroll
    for (int h = 0; h < HEADS; ++h) {
        float s = 0.0f;
        #pragma unroll
        for (int sc = 0; sc < SC; ++sc)
            s += c_part[(((size_t)b * SC + sc) * HEADS + h) * D2 + d];
        cvec[((size_t)b * HEADS + h) * D2 + d] = s;
    }
}

// ---------------------------------------------------------------------------
// 5) att[b,h,u] = sum_d c[b,h,d]*vk[d,h,u]; out[b,u] = sum_f att[f]*sw[f,u]
//    grid(B), block(128)
// ---------------------------------------------------------------------------
__global__ void out_kernel(const float* __restrict__ cvec,
                           const float* __restrict__ vk,
                           const float* __restrict__ swk,
                           float* __restrict__ out) {
    const int b = blockIdx.x;
    const int tid = threadIdx.x; // 0..127

    __shared__ float cl[HEADS * D2]; // 1024 floats
    __shared__ float att[F];         // 512 floats

    const float* cb = cvec + (size_t)b * HEADS * D2;
    #pragma unroll
    for (int i = 0; i < 8; ++i) cl[tid + i * 128] = cb[tid + i * 128];
    __syncthreads();

    #pragma unroll
    for (int i = 0; i < 4; ++i) {
        const int f = tid + i * 128;
        const int h = f >> 7;
        const int u = f & 127;
        float a = 0.0f;
        #pragma unroll 4
        for (int d = 0; d < D2; ++d)
            a += cl[h * D2 + d] * vk[(size_t)d * F + h * UNITS + u];
        att[f] = a;
    }
    __syncthreads();

    float o = 0.0f;
    #pragma unroll 4
    for (int f = 0; f < F; ++f)
        o += att[f] * swk[(size_t)f * UNITS + tid];
    out[(size_t)b * UNITS + tid] = o;
}

// ---------------------------------------------------------------------------
extern "C" void kernel_launch(void* const* d_in, const int* in_sizes, int n_in,
                              void* d_out, int out_size, void* d_ws, size_t ws_size,
                              hipStream_t stream) {
    const float* x   = (const float*)d_in[0];
    const float* qk  = (const float*)d_in[1];
    const float* kk  = (const float*)d_in[2];
    const float* vk  = (const float*)d_in[3];
    const float* swk = (const float*)d_in[4];
    float* out = (float*)d_out;

    // Workspace carve-up (floats), ~36 MB total:
    float* seq    = (float*)d_ws;                       // B*T*2D        = 8,388,608
    float* wkp    = seq    + (size_t)B * T * D2;        // B*16*256      =   131,072
    float* sco    = wkp    + (size_t)B * HP * D2;       // B*4*1024      =   131,072
    float* cvec   = sco    + (size_t)B * HEADS * T;     // B*4*256       =    32,768
    float* locend = cvec   + (size_t)B * HEADS * D2;    // B*CCH*D       =    65,536
    float* init   = locend + (size_t)B * CCH * D;       // B*CCH*D       =    65,536
    float* probs  = init   + (size_t)B * CCH * D;       // B*4*1024      =   131,072
    float* c_part = probs  + (size_t)B * HEADS * T;     // B*SC*4*256    =   262,144

    scan_partial_kernel<<<B * CCH, D,   0, stream>>>(x, locend);
    scan_combine_kernel<<<B,       D,   0, stream>>>(locend, init);
    scan_write_kernel  <<<B * CCH, D,   0, stream>>>(x, init, seq);
    qwk_kernel         <<<B * HP,  128, 0, stream>>>(seq, qk, kk, wkp);
    score_wmma_kernel  <<<B * 8,   32,  0, stream>>>(seq, wkp, sco);
    softmax_kernel     <<<B,       256, 0, stream>>>(sco, probs);
    c_partial_kernel   <<<B * SC,  256, 0, stream>>>(seq, probs, c_part);
    c_reduce_kernel    <<<B,       256, 0, stream>>>(c_part, cvec);
    out_kernel         <<<B,       128, 0, stream>>>(cvec, vk, swk, out);
}
